// SIRcell_1185410974017
// MI455X (gfx1250) — matrix operand order repaired and checked by reading, not compiled, not run on hardware
//
#include <hip/hip_runtime.h>
#include <hip/hip_bf16.h>

// ---------------- problem constants ----------------
#define BB 4
#define NN 5000
#define TI 128          // rows per block
#define TJ 128          // cols per tile
#define LDSTR 136       // f16 row stride in LDS tile (128 + 8 pad, 16B-aligned rows)

typedef _Float16 v2h  __attribute__((ext_vector_type(2)));
typedef _Float16 v4h  __attribute__((ext_vector_type(4)));
typedef _Float16 v8h  __attribute__((ext_vector_type(8)));
typedef _Float16 v16h __attribute__((ext_vector_type(16)));
typedef __fp16   f16x2 __attribute__((ext_vector_type(2)));
typedef __fp16   f16x4 __attribute__((ext_vector_type(4)));
typedef float    v4f  __attribute__((ext_vector_type(4)));
typedef float    v8f  __attribute__((ext_vector_type(8)));

// ---------------- kernel 0: zero the column accumulator ----------------
__global__ void sir_zero(float* __restrict__ tc) {
    int i = blockIdx.x * blockDim.x + threadIdx.x;
    if (i < BB * NN) tc[i] = 0.0f;
}

// ---------------- kernel 1: single pass over mob ----------------
// tr[b,i] = sum_j mob[b,i,j] * Iv[b,j]            (WMMA f16xf16->f32, owner write)
// tc[b,j] += sum_i mob[b,i,j] * Iv[b,i]/pop[b,i]  (fused into staging, full f32)
__global__ __launch_bounds__(256) void sir_stream(const float* __restrict__ mob,
                                                  const float* __restrict__ SIR,
                                                  float* __restrict__ tc,
                                                  float* __restrict__ tr) {
    __shared__ __align__(16) _Float16 tile[TI * LDSTR];   // f16 mob tile (WMMA only)
    __shared__ __align__(16) _Float16 ivh[TJ];            // f16 Iv for this col tile
    __shared__ float wrow[TI];                            // f32 Iv/pop for block rows
    __shared__ __align__(16) float colred[8][32][4];      // [wave][c4][q] partials

    const int b    = blockIdx.y;
    const int i0   = blockIdx.x * TI;
    const int t    = threadIdx.x;
    const int lane = t & 31;            // wave32
    const int wv   = t >> 5;            // 0..7

    const float* mobB = mob + (size_t)b * NN * NN;
    const float* sirB = SIR + (size_t)b * NN * 3;

    // stage per-row weights w = I / (S+I+R) once per block
    if (t < TI) {
        int i = i0 + t;
        float w = 0.0f;
        if (i < NN) {
            float S = sirB[i * 3 + 0], I = sirB[i * 3 + 1], R = sirB[i * 3 + 2];
            w = I / (S + I + R);
        }
        wrow[t] = w;
    }
    __syncthreads();

    // each thread stages rows r = k*8 + wv (k = 0..15): hoist its 16 w values
    float wreg[16];
#pragma unroll
    for (int k = 0; k < 16; ++k) wreg[k] = wrow[k * 8 + wv];

    // term_row WMMA accumulator: wave wv owns rows i0 + 16*wv .. +15
    v8f acc;
#pragma unroll
    for (int z = 0; z < 8; ++z) acc[z] = 0.0f;

    const int c4    = t & 31;             // this thread's float4-column (fixed!)
    const int nrow  = lane & 15;          // B operand: row within wave's 16-row strip
    const int khalf = (lane >> 4) << 3;   // unused; A and B both take K base from bcol
    const int bcol  = (lane >> 4) << 4;   // B operand: K chunk base (0 or 16)

    for (int j0 = 0; j0 < NN; j0 += TJ) {
        __syncthreads();  // previous iteration's LDS readers are done

        // ---- stage Iv (= I column vector) as f16 ----
        if (t < TJ) {
            int j = j0 + t;
            float iv = (j < NN) ? sirB[j * 3 + 1] : 0.0f;
            ivh[t] = (_Float16)iv;
        }

        // ---- stage mob tile (f32 -> f16 LDS) and FUSE the column GEMV in f32 ----
        v4f cacc = {0.0f, 0.0f, 0.0f, 0.0f};
#pragma unroll
        for (int k = 0; k < 16; ++k) {
            int r  = k * 8 + wv;
            int gi = i0 + r, gj = j0 + c4 * 4;
            v4f v = {0.0f, 0.0f, 0.0f, 0.0f};
            if (gi < NN && gj < NN)   // NN%4==0 -> no straddle
                v = __builtin_nontemporal_load((const v4f*)(mobB + (size_t)gi * NN + gj));
            cacc += v * wreg[k];                               // 2x v_pk_fma_f32
            f16x2 p01 = __builtin_amdgcn_cvt_pkrtz(v[0], v[1]);  // v_cvt_pk_rtz_f16_f32
            f16x2 p23 = __builtin_amdgcn_cvt_pkrtz(v[2], v[3]);
            f16x4 h = __builtin_shufflevector(p01, p23, 0, 1, 2, 3);
            *(f16x4*)(&tile[r * LDSTR + c4 * 4]) = h;   // bit-identical IEEE f16
        }
        *(v4f*)(&colred[wv][c4][0]) = cacc;   // per-thread slot, no race
        __syncthreads();

        // ---- term_row: 4 sub-tiles of 16x32 per wave, v_wmma_f32_16x16x32_f16 ----
        // B[k,n] = M[i0 + 16*wv + n, j0 + cj + k]  (lane-contiguous row fragments)
        // A[m,k] = Iv[j0 + cj + k]                 (broadcast over m)
#pragma unroll
        for (int s = 0; s < TJ / 32; ++s) {
            int cj = s * 32;
            const _Float16* bp = &tile[(wv * 16 + nrow) * LDSTR + cj + bcol];
            v8h b_lo = *(const v8h*)(bp);
            v8h b_hi = *(const v8h*)(bp + 8);
            const _Float16* ap = &ivh[cj + bcol];
            v8h a_lo = *(const v8h*)(ap);
            v8h a_hi = *(const v8h*)(ap + 8);
            v16h A = __builtin_shufflevector(a_lo, a_hi, 0, 1, 2, 3, 4, 5, 6, 7,
                                             8, 9, 10, 11, 12, 13, 14, 15);
            v16h Bm = __builtin_shufflevector(b_lo, b_hi, 0, 1, 2, 3, 4, 5, 6, 7,
                                              8, 9, 10, 11, 12, 13, 14, 15);
            acc = __builtin_amdgcn_wmma_f32_16x16x32_f16(false, A, false, Bm,
                                                         (short)0, acc, false, false);
        }

        // ---- combine column partials across the 8 waves, one atomic per column ----
        if (t < TJ) {
            int cc4 = t >> 2, q = t & 3;
            float s = 0.0f;
#pragma unroll
            for (int g = 0; g < 8; ++g) s += colred[g][cc4][q];
            int j = j0 + t;
            if (j < NN) atomicAdd(&tc[b * NN + j], s);
        }
    }

    // ---- write term_row numerator (rows owned exclusively by this block/wave) ----
    // D layout: every VGPR of acc holds tr[row = 16*wv + (lane&15)]
    {
        int i = i0 + wv * 16 + (lane & 15);
        if (lane < 16 && i < NN) tr[b * NN + i] = acc[0];
    }
}

// ---------------- kernel 2: elementwise SIR update ----------------
__global__ void sir_finalize(const float* __restrict__ SIR,
                             const float* __restrict__ pb,
                             const float* __restrict__ pg,
                             const float* __restrict__ tc,
                             const float* __restrict__ tr,
                             float* __restrict__ out) {
    int idx = blockIdx.x * blockDim.x + threadIdx.x;   // over B*N
    if (idx >= BB * NN) return;
    int b = idx / NN;
    float S = SIR[idx * 3 + 0], I = SIR[idx * 3 + 1], R = SIR[idx * 3 + 2];
    float pop      = S + I + R;
    float term_row = tr[idx] / pop;
    float prop     = tc[idx] + term_row;
    float I_new    = pb[b] * prop;
    float Ig       = I * pg[b];
    out[idx * 4 + 0] = I_new;           // I_new
    out[idx * 4 + 1] = S - I_new;       // S_t
    out[idx * 4 + 2] = I + I_new - Ig;  // I_t
    out[idx * 4 + 3] = Ig + R;          // R_t
}

// ---------------- launch ----------------
extern "C" void kernel_launch(void* const* d_in, const int* in_sizes, int n_in,
                              void* d_out, int out_size, void* d_ws, size_t ws_size,
                              hipStream_t stream) {
    const float* pb  = (const float*)d_in[0];   // (B,1,1)
    const float* pg  = (const float*)d_in[1];   // (B,1,1)
    const float* mob = (const float*)d_in[2];   // (B,N,N)
    const float* SIR = (const float*)d_in[3];   // (B,N,3)
    float* out = (float*)d_out;                 // (B,N,4)

    float* tc = (float*)d_ws;                   // B*N col accumulator
    float* tr = tc + BB * NN;                   // B*N row numerators (160 KB total)

    sir_zero<<<(BB * NN + 255) / 256, 256, 0, stream>>>(tc);

    dim3 grid((NN + TI - 1) / TI, BB);          // 40 x 4 blocks, 256 thr (8 waves)
    sir_stream<<<grid, 256, 0, stream>>>(mob, SIR, tc, tr);

    sir_finalize<<<(BB * NN + 255) / 256, 256, 0, stream>>>(SIR, pb, pg, tc, tr, out);
}
